// BailingMoeV2SparseMoeBlock_7224134992006
// MI455X (gfx1250) — matrix operand order repaired and checked by reading, not compiled, not run on hardware
//
#include <hip/hip_runtime.h>
#include <hip/hip_bf16.h>
#include <math.h>

typedef __attribute__((ext_vector_type(16))) __bf16 v16bf;
typedef __attribute__((ext_vector_type(2)))  __bf16 v2bf;
typedef __attribute__((ext_vector_type(8)))  float  v8f;
typedef int v4i_gcc __attribute__((vector_size(16)));   // matches builtin param

#define T_TOK 1024
#define D_DIM 2048
#define E_EXP 64
#define F_DIM 512
#define TOPK  8
#define NGRP  8
#define GSZ   8
#define TOPKG 4
#define RSCALE 2.5f
#define CAP   1024
#define NE    65   // 64 routed + 1 shared pseudo-expert
#define WPAD  40   // padded LDS row (halves): 80B rows, keeps 16B alignment

// workspace layout (bytes)
#define OFF_COUNTS 0
#define OFF_TOKID  4096
#define OFF_TOKW   (OFF_TOKID + NE*CAP*4)
#define OFF_ACT    (OFF_TOKW  + NE*CAP*4)

struct U32x8 { uint4 lo, hi; };

__device__ __forceinline__ v8f wmma_bf16(v16bf a, v16bf b, v8f c) {
  return __builtin_amdgcn_wmma_f32_16x16x32_bf16(
      false, a, false, b, (short)0, c, false, false);
}

__device__ __forceinline__ unsigned int pk_bf16(float a, float b) {
#if __has_builtin(__builtin_amdgcn_cvt_pk_bf16_f32)
  return __builtin_bit_cast(unsigned int, __builtin_amdgcn_cvt_pk_bf16_f32(a, b));
#else
  v2bf v;
  v[0] = (__bf16)a;
  v[1] = (__bf16)b;
  return __builtin_bit_cast(unsigned int, v);
#endif
}

__device__ __forceinline__ v16bf frag_from_lds2(const __bf16* p0, const __bf16* p1) {
  U32x8 u;
  u.lo = *(const uint4*)p0;
  u.hi = *(const uint4*)p1;
  return __builtin_bit_cast(v16bf, u);
}

// ---- gfx1250 async global->LDS copy (ASYNCcnt), with safe fallback --------
#if __has_builtin(__builtin_amdgcn_global_load_async_to_lds_b128)
#define HAVE_ASYNC_LDS 1
__device__ __forceinline__ void async_copy_b128(const void* g, void* l) {
  __builtin_amdgcn_global_load_async_to_lds_b128(
      (v4i_gcc*)(void*)(uintptr_t)g, (v4i_gcc*)l, 0, 0);
}
__device__ __forceinline__ void wait_async0() {
#if __has_builtin(__builtin_amdgcn_s_wait_asynccnt)
  __builtin_amdgcn_s_wait_asynccnt(0);
#else
  asm volatile("s_wait_asynccnt 0x0" ::: "memory");
#endif
}
#else
__device__ __forceinline__ void wait_async0() {}
#endif

// ---------------- init: zero counts, build shared-expert token list ---------
__global__ void moe_init_kernel(int* __restrict__ counts,
                                int* __restrict__ tok_ids,
                                float* __restrict__ tok_w) {
  int i = blockIdx.x * blockDim.x + threadIdx.x;
  if (i < NE) counts[i] = (i == E_EXP) ? T_TOK : 0;
  if (i < T_TOK) {
    tok_ids[E_EXP * CAP + i] = i;
    tok_w [E_EXP * CAP + i] = 1.0f;
  }
}

// ---------------- gate: logits, sigmoid, group-limited top-k, scatter -------
__global__ void moe_gate_kernel(const float* __restrict__ x,
                                const float* __restrict__ gate_w,
                                const float* __restrict__ expert_bias,
                                int* __restrict__ counts,
                                int* __restrict__ tok_ids,
                                float* __restrict__ tok_w) {
  const int t = blockIdx.x;        // one block per token, 64 threads
  const int e = threadIdx.x;       // one expert per thread
  __shared__ float s_score[E_EXP];
  __shared__ float s_sr[E_EXP];

  const float4* xr = (const float4*)(x + (size_t)t * D_DIM);
  const float4* wr = (const float4*)(gate_w + (size_t)e * D_DIM);
  float acc = 0.f;
  for (int d = 0; d < D_DIM / 4; ++d) {
    float4 xv = xr[d], wv = wr[d];
    acc += xv.x * wv.x + xv.y * wv.y + xv.z * wv.z + xv.w * wv.w;
  }
  float sc = 1.0f / (1.0f + __expf(-acc));   // sigmoid
  s_score[e] = sc;
  s_sr[e]    = sc + expert_bias[e];
  __syncthreads();

  if (e == 0) {
    float gsc[NGRP];
    for (int g = 0; g < NGRP; ++g) {
      float m1 = -3e38f, m2 = -3e38f;
      for (int j = 0; j < GSZ; ++j) {
        float v = s_sr[g * GSZ + j];
        if (v > m1) { m2 = m1; m1 = v; } else if (v > m2) { m2 = v; }
      }
      gsc[g] = m1 + m2;
    }
    bool gsel[NGRP];
    for (int g = 0; g < NGRP; ++g) gsel[g] = false;
    for (int it = 0; it < TOPKG; ++it) {
      int best = -1; float bv = -3e38f;
      for (int g = 0; g < NGRP; ++g)
        if (!gsel[g] && gsc[g] > bv) { bv = gsc[g]; best = g; }
      gsel[best] = true;
    }
    int  idx[TOPK];
    bool esel[E_EXP];
    for (int i = 0; i < E_EXP; ++i) esel[i] = false;
    float wsum = 0.f;
    for (int it = 0; it < TOPK; ++it) {
      int best = -1; float bv = -3e38f;
      for (int q = 0; q < E_EXP; ++q) {
        if (esel[q] || !gsel[q / GSZ]) continue;
        float v = s_sr[q];
        if (v > bv) { bv = v; best = q; }
      }
      esel[best] = true; idx[it] = best; wsum += s_score[best];
    }
    float inv = RSCALE / (wsum + 1e-20f);
    for (int kk = 0; kk < TOPK; ++kk) {
      int q = idx[kk];
      int slot = atomicAdd(&counts[q], 1);
      tok_ids[q * CAP + slot] = t;
      tok_w [q * CAP + slot] = s_score[q] * inv;
    }
  }
}

// ------------- stage1: act[e,slot,0:512] = silu(x@Wg) * (x@Wu), bf16 --------
// grid: (expert 0..64, mtile 0..15 [64 tokens], nblk 0..3 [128 act cols])
__global__ void moe_stage1_kernel(const float* __restrict__ x,
                                  const float* __restrict__ w_gate_up,
                                  const float* __restrict__ shared_w_gate_up,
                                  const int* __restrict__ counts,
                                  const int* __restrict__ tok_ids,
                                  __bf16* __restrict__ act) {
  const int e  = blockIdx.x;
  const int cnt = counts[e];
  const int m0 = blockIdx.y * 64;
  if (m0 >= cnt) return;
  const int gbase = blockIdx.z * 128;           // g col base; u col = +512
  const float* wgu = (e < E_EXP) ? (w_gate_up + (size_t)e * D_DIM * (2 * F_DIM))
                                 : shared_w_gate_up;

  __shared__ __bf16 Xs[2][64][WPAD];    // [buf][row][k]  (A is K-contiguous)
  __shared__ __bf16 Ws[2][256][WPAD];   // [buf][c][k]    (B is K-contiguous)
  __shared__ int s_tok[64];

  const int tid  = threadIdx.x;
  const int lane = tid & 31;
  const int wv   = tid >> 5;        // 8 waves
  const int mtile = wv & 3;         // 4 M-tiles of 16 rows
  const int half  = wv >> 2;        // 2 halves of 64 cols

  if (tid < 64) {
    int slot = m0 + tid;
    s_tok[tid] = (slot < cnt) ? tok_ids[e * CAP + slot] : 0;
  }
  __syncthreads();

  v8f accG[4], accU[4];
  for (int nt = 0; nt < 4; ++nt)
    for (int r = 0; r < 8; ++r) { accG[nt][r] = 0.f; accU[nt][r] = 0.f; }

  const int rowA = mtile * 16 + (lane & 15);
  const int kbA  = (lane >> 4) * 16;
  const int kbB  = (lane >> 4) * 16;
  const int lN   = lane & 15;

  // staging roles
  const int xr_ = tid >> 2, xk_ = (tid & 3) * 8;     // X: 8 halves per thread
  const int wc_ = tid;                                // W: one column per thread
  const int wcol_ = (wc_ < 128) ? (gbase + wc_) : (F_DIM + gbase + (wc_ - 128));
  const float* wcolp = wgu + wcol_;                   // column base (stride 2F)
  const float* xrowp = x + (size_t)s_tok[xr_] * D_DIM + xk_;

  auto stage_tiles = [&](int buf, int k0) {
    // X tile: 2x float4 -> 1x b128 store
    float4 a4 = *(const float4*)(xrowp + k0);
    float4 b4 = *(const float4*)(xrowp + k0 + 4);
    uint4 px = { pk_bf16(a4.x, a4.y), pk_bf16(a4.z, a4.w),
                 pk_bf16(b4.x, b4.y), pk_bf16(b4.z, b4.w) };
    *(uint4*)&Xs[buf][xr_][xk_] = px;
    // W tile transposed: 32 dword loads -> 4x b128 stores
    float f[32];
#pragma unroll
    for (int r = 0; r < 32; ++r)
      f[r] = wcolp[(size_t)(k0 + r) * (2 * F_DIM)];
#pragma unroll
    for (int q = 0; q < 4; ++q) {
      uint4 p = { pk_bf16(f[q*8+0], f[q*8+1]), pk_bf16(f[q*8+2], f[q*8+3]),
                  pk_bf16(f[q*8+4], f[q*8+5]), pk_bf16(f[q*8+6], f[q*8+7]) };
      *(uint4*)&Ws[buf][wc_][q * 8] = p;
    }
  };

  stage_tiles(0, 0);
  __syncthreads();

  for (int it = 0; it < D_DIM / 32; ++it) {
    const int k0 = it * 32;
    const int cur = it & 1;
    if (k0 + 32 < D_DIM) {
      if (k0 + 64 < D_DIM)
        __builtin_prefetch(&wcolp[(size_t)(k0 + 64) * (2 * F_DIM)], 0, 1);
      stage_tiles(cur ^ 1, k0 + 32);    // overlaps with WMMA below
    }

    // preload fragments, then burst the 8 WMMAs
    v16bf a = frag_from_lds2(&Xs[cur][rowA][kbA], &Xs[cur][rowA][kbA + 8]);
    v16bf bg[4], bu[4];
#pragma unroll
    for (int nt = 0; nt < 4; ++nt) {
      int cb = half * 64 + nt * 16 + lN;
      bg[nt] = frag_from_lds2(&Ws[cur][cb][kbB],       &Ws[cur][cb][kbB + 8]);
      bu[nt] = frag_from_lds2(&Ws[cur][128 + cb][kbB], &Ws[cur][128 + cb][kbB + 8]);
    }
#pragma unroll
    for (int nt = 0; nt < 4; ++nt) {
      accG[nt] = wmma_bf16(a, bg[nt], accG[nt]);
      accU[nt] = wmma_bf16(a, bu[nt], accU[nt]);
    }
    __syncthreads();
  }

  // epilogue: act = silu(g) * u
  for (int nt = 0; nt < 4; ++nt) {
    for (int r = 0; r < 8; ++r) {
      int row  = mtile * 16 + r + (lane >> 4) * 8;
      int slot = m0 + row;
      if (slot >= cnt) continue;
      float g = accG[nt][r], u = accU[nt][r];
      float s = g * u / (1.0f + __expf(-g));
      int col = gbase + half * 64 + nt * 16 + lN;
      act[((size_t)e * CAP + slot) * F_DIM + col] = (__bf16)s;
    }
  }
}

// ------------- stage2: out[t, :] += w * (act @ w_down) ----------------------
// grid: (expert 0..64, mtile 0..15 [64 tokens], nblk 0..7 [256 out cols])
__global__ void moe_stage2_kernel(const __bf16* __restrict__ act,
                                  const float* __restrict__ w_down,
                                  const float* __restrict__ shared_w_down,
                                  const int* __restrict__ counts,
                                  const int* __restrict__ tok_ids,
                                  const float* __restrict__ tok_w,
                                  float* __restrict__ out) {
  const int e  = blockIdx.x;
  const int cnt = counts[e];
  const int m0 = blockIdx.y * 64;
  if (m0 >= cnt) return;
  const int nbase = blockIdx.z * 256;
  const float* wd = (e < E_EXP) ? (w_down + (size_t)e * F_DIM * D_DIM)
                                : shared_w_down;

  __shared__ __bf16 As[2][64][WPAD];
  __shared__ __bf16 Ws[2][256][WPAD];
  __shared__ int   s_tok[64];
  __shared__ float s_w[64];

  const int tid  = threadIdx.x;
  const int lane = tid & 31;
  const int wv   = tid >> 5;
  const int mtile = wv & 3;
  const int half  = wv >> 2;        // 2 halves of 128 out-cols

  if (tid < 64) {
    int slot = m0 + tid;
    bool ok = slot < cnt;
    s_tok[tid] = ok ? tok_ids[e * CAP + slot] : 0;
    s_w[tid]   = ok ? tok_w [e * CAP + slot] : 0.f;
  }

  v8f acc[8];
  for (int nt = 0; nt < 8; ++nt)
    for (int r = 0; r < 8; ++r) acc[nt][r] = 0.f;

  const int rowA = mtile * 16 + (lane & 15);
  const int kbA  = (lane >> 4) * 16;
  const int kbB  = (lane >> 4) * 16;
  const int lN   = lane & 15;

  const int  ar_ = tid >> 2, ak_ = (tid & 3) * 8;
  const bool arok = (m0 + ar_) < cnt;
  const __bf16* arowp = act + ((size_t)e * CAP + (m0 + ar_)) * F_DIM + ak_;
  const int wc_ = tid;
  const float* wcolp = wd + nbase + wc_;              // column base (stride D)

  // pre-zero masked A rows once (async path never rewrites them)
  if (!arok) {
    uint4 z = {0u, 0u, 0u, 0u};
    *(uint4*)&As[0][ar_][ak_] = z;
    *(uint4*)&As[1][ar_][ak_] = z;
  }
  __syncthreads();

  auto stage_tiles = [&](int buf, int k0) {
    // ACT tile (already bf16): async global->LDS b128 when available
#ifdef HAVE_ASYNC_LDS
    if (arok) async_copy_b128(arowp + k0, &As[buf][ar_][ak_]);
#else
    if (arok) *(uint4*)&As[buf][ar_][ak_] = *(const uint4*)(arowp + k0);
#endif
    // W tile transposed: 32 dword loads -> 4x b128 stores
    float f[32];
#pragma unroll
    for (int r = 0; r < 32; ++r)
      f[r] = wcolp[(size_t)(k0 + r) * D_DIM];
#pragma unroll
    for (int q = 0; q < 4; ++q) {
      uint4 p = { pk_bf16(f[q*8+0], f[q*8+1]), pk_bf16(f[q*8+2], f[q*8+3]),
                  pk_bf16(f[q*8+4], f[q*8+5]), pk_bf16(f[q*8+6], f[q*8+7]) };
      *(uint4*)&Ws[buf][wc_][q * 8] = p;
    }
  };

  stage_tiles(0, 0);
  wait_async0();
  __syncthreads();

  for (int it = 0; it < F_DIM / 32; ++it) {
    const int k0 = it * 32;
    const int cur = it & 1;
    if (k0 + 32 < F_DIM) {
      if (k0 + 64 < F_DIM)
        __builtin_prefetch(&wcolp[(size_t)(k0 + 64) * D_DIM], 0, 1);
      stage_tiles(cur ^ 1, k0 + 32);
    }

    v16bf a = frag_from_lds2(&As[cur][rowA][kbA], &As[cur][rowA][kbA + 8]);
    v16bf b[8];
#pragma unroll
    for (int nt = 0; nt < 8; ++nt) {
      int cb = half * 128 + nt * 16 + lN;
      b[nt] = frag_from_lds2(&Ws[cur][cb][kbB], &Ws[cur][cb][kbB + 8]);
    }
#pragma unroll
    for (int nt = 0; nt < 8; ++nt)
      acc[nt] = wmma_bf16(a, b[nt], acc[nt]);

    wait_async0();      // async copies for buf cur^1 landed before barrier
    __syncthreads();
  }

  for (int nt = 0; nt < 8; ++nt) {
    for (int r = 0; r < 8; ++r) {
      int row  = mtile * 16 + r + (lane >> 4) * 8;
      int slot = m0 + row;
      if (slot >= cnt) continue;
      int col = nbase + half * 128 + nt * 16 + lN;
      float v = acc[nt][r] * s_w[row];
      unsafeAtomicAdd(&out[(size_t)s_tok[row] * D_DIM + col], v);
    }
  }
}

extern "C" void kernel_launch(void* const* d_in, const int* in_sizes, int n_in,
                              void* d_out, int out_size, void* d_ws, size_t ws_size,
                              hipStream_t stream) {
  const float* x      = (const float*)d_in[0];   // [T, D]
  const float* gate_w = (const float*)d_in[1];   // [E, D]
  const float* ebias  = (const float*)d_in[2];   // [E]
  const float* wgu    = (const float*)d_in[3];   // [E, D, 2F]
  const float* wdn    = (const float*)d_in[4];   // [E, F, D]
  const float* swgu   = (const float*)d_in[5];   // [D, 2F]
  const float* swdn   = (const float*)d_in[6];   // [F, D]
  float* out = (float*)d_out;

  char* ws = (char*)d_ws;
  int*    counts  = (int*)  (ws + OFF_COUNTS);
  int*    tok_ids = (int*)  (ws + OFF_TOKID);
  float*  tok_w   = (float*)(ws + OFF_TOKW);
  __bf16* act     = (__bf16*)(ws + OFF_ACT);

  (void)hipMemsetAsync(d_out, 0, (size_t)T_TOK * D_DIM * sizeof(float), stream);
  moe_init_kernel<<<(T_TOK + 255) / 256, 256, 0, stream>>>(counts, tok_ids, tok_w);
  moe_gate_kernel<<<T_TOK, 64, 0, stream>>>(x, gate_w, ebias, counts, tok_ids, tok_w);
  moe_stage1_kernel<<<dim3(NE, CAP / 64, 4), 256, 0, stream>>>(
      x, wgu, swgu, counts, tok_ids, act);
  moe_stage2_kernel<<<dim3(NE, CAP / 64, 8), 256, 0, stream>>>(
      act, wdn, swdn, counts, tok_ids, tok_w, out);
}
